// GAT_36223754174564
// MI455X (gfx1250) — matrix operand order repaired and checked by reading, not compiled, not run on hardware
//
#include <hip/hip_runtime.h>
#include <hip/hip_bf16.h>
#include <math.h>

// ---------------------------------------------------------------- types
typedef __attribute__((ext_vector_type(16))) __bf16          v16bf;
typedef __attribute__((ext_vector_type(8)))  float           v8f;
typedef __attribute__((ext_vector_type(8)))  unsigned short  ushort8;
typedef __attribute__((ext_vector_type(4)))  int             v4i;

union BF16Frag { ushort8 u[2]; v16bf v; };

#define NCOL 256      // HEADS*HID
#define HEADS 8
#define HID 32

#if __has_builtin(__builtin_amdgcn_global_load_async_to_lds_b128)
#define HAVE_ASYNC_LDS 1
#else
#define HAVE_ASYNC_LDS 0
#endif

__device__ __forceinline__ unsigned short f2bf(float f) {
    unsigned u = __float_as_uint(f);
    unsigned r = u + 0x7FFFu + ((u >> 16) & 1u);   // round-to-nearest-even
    return (unsigned short)(r >> 16);
}

__device__ __forceinline__ void atomicMaxF(float* addr, float val) {
    if (val >= 0.0f) atomicMax((int*)addr, __float_as_int(val));
    else             atomicMin((unsigned int*)addr, __float_as_uint(val));
}

// 16-byte global->LDS copy: async on CDNA5 (ASYNCcnt), plain fallback otherwise
__device__ __forceinline__ void cp16(const unsigned short* g, unsigned short* l) {
#if HAVE_ASYNC_LDS
    __builtin_amdgcn_global_load_async_to_lds_b128(
        (__attribute__((address_space(1))) v4i*)g,
        (__attribute__((address_space(3))) v4i*)l, 0, 0);
#else
    *(ushort8*)l = *(const ushort8*)g;
#endif
}

// ---------------------------------------------------------------- conversions
// fp32 [nRows x K] -> bf16 [Mpad x K], zero pad rows
__global__ void convert_in_kernel(const float* __restrict__ src, unsigned short* __restrict__ dst,
                                  int nRows, int kshift, long long total) {
    long long idx = (long long)blockIdx.x * blockDim.x + threadIdx.x;
    if (idx >= total) return;
    int r = (int)(idx >> kshift);
    int c = (int)(idx & ((1 << kshift) - 1));
    float v = (r < nRows) ? src[(size_t)r * (1 << kshift) + c] : 0.0f;
    dst[idx] = f2bf(v);
}

// fp32 W [K x 256] -> bf16 Wt [256 x K] (transposed)
__global__ void convert_wT_kernel(const float* __restrict__ W, unsigned short* __restrict__ Wt,
                                  int kshift, int total) {
    int idx = blockIdx.x * blockDim.x + threadIdx.x;
    if (idx >= total) return;
    int n = idx >> kshift;
    int k = idx & ((1 << kshift) - 1);
    Wt[idx] = f2bf(W[(size_t)k * NCOL + n]);
}

// ---------------------------------------------------------------- GEMM (WMMA bf16)
// H[Mpad x 256] = A[Mpad x K](bf16) * Bt[256 x K](bf16, pre-transposed).
// Block tile 128x128, 8 waves 4(M)x2(N), wave tile 32x64 = 2x4 WMMA 16x16x32.
// Double-buffered LDS filled by async global->LDS b128 copies (4 per wave/tile).
__global__ void __launch_bounds__(256)
gat_gemm_wmma(const unsigned short* __restrict__ A, const unsigned short* __restrict__ Bt,
              float* __restrict__ H, int K) {
    __shared__ __align__(16) unsigned short As[2][128 * 32];  // [m][k], 8KB each
    __shared__ __align__(16) unsigned short Bs[2][128 * 32];  // [n][k], 8KB each

    const int t    = threadIdx.x;
    const int lane = t & 31;
    const int wave = t >> 5;
    const int wm   = wave >> 1;          // 0..3  (M, 32 rows each)
    const int wn   = wave & 1;           // 0..1  (N, 64 cols each)
    const int half = lane >> 4;          // 0/1
    const int l16  = lane & 15;

    const int blockM = blockIdx.x * 128;
    const int blockN = blockIdx.y * 128;

    // chunk decomposition for the two tiles: 512 chunks of 16B each
    const int r0 = t >> 2,          c0 = (t & 3) * 8;          // chunks t
    const int r1 = (t + 256) >> 2,  c1 = ((t + 256) & 3) * 8;  // chunks t+256

    v8f acc[2][4] = {};

    const int steps = K >> 5;
    // prologue: stage tile 0 into buffer 0
    {
        cp16(A  + (size_t)(blockM + r0) * K + c0, &As[0][r0 * 32 + c0]);
        cp16(A  + (size_t)(blockM + r1) * K + c1, &As[0][r1 * 32 + c1]);
        cp16(Bt + (size_t)(blockN + r0) * K + c0, &Bs[0][r0 * 32 + c0]);
        cp16(Bt + (size_t)(blockN + r1) * K + c1, &Bs[0][r1 * 32 + c1]);
    }

    for (int s = 0; s < steps; ++s) {
        const int cur = s & 1;
        if (s + 1 < steps) {
            const int k0 = (s + 1) << 5;
            cp16(A  + (size_t)(blockM + r0) * K + k0 + c0, &As[cur ^ 1][r0 * 32 + c0]);
            cp16(A  + (size_t)(blockM + r1) * K + k0 + c1, &As[cur ^ 1][r1 * 32 + c1]);
            cp16(Bt + (size_t)(blockN + r0) * K + k0 + c0, &Bs[cur ^ 1][r0 * 32 + c0]);
            cp16(Bt + (size_t)(blockN + r1) * K + k0 + c1, &Bs[cur ^ 1][r1 * 32 + c1]);
#if HAVE_ASYNC_LDS
            asm volatile("s_wait_asynccnt 0x4" ::: "memory");  // current tile's 4 done (in-order)
#endif
        } else {
#if HAVE_ASYNC_LDS
            asm volatile("s_wait_asynccnt 0x0" ::: "memory");
#endif
        }
        __syncthreads();

        // A fragments: 16x32 bf16; lane(half,l16): two contiguous 16B runs per lane
        BF16Frag afr[2], bfr[4];
        #pragma unroll
        for (int mi = 0; mi < 2; ++mi) {
            int r = wm * 32 + mi * 16 + l16;
            afr[mi].u[0] = *(const ushort8*)&As[cur][r * 32 + half * 8];
            afr[mi].u[1] = *(const ushort8*)&As[cur][r * 32 + 16 + half * 8];
        }
        // B fragments: 32x16 bf16; lane(half,N=l16): two contiguous 16B runs per lane
        #pragma unroll
        for (int ni = 0; ni < 4; ++ni) {
            int cn = wn * 64 + ni * 16 + l16;
            bfr[ni].u[0] = *(const ushort8*)&Bs[cur][cn * 32 + half * 8];
            bfr[ni].u[1] = *(const ushort8*)&Bs[cur][cn * 32 + 16 + half * 8];
        }
        #pragma unroll
        for (int mi = 0; mi < 2; ++mi)
            #pragma unroll
            for (int ni = 0; ni < 4; ++ni)
                acc[mi][ni] = __builtin_amdgcn_wmma_f32_16x16x32_bf16(
                    false, afr[mi].v, false, bfr[ni].v,
                    (short)0, acc[mi][ni], false, false);
        __syncthreads();
    }

    // D layout: lane(half,N=l16), VGPR j -> row M = half*8 + j.  Unguarded (Mpad rows).
    #pragma unroll
    for (int mi = 0; mi < 2; ++mi) {
        #pragma unroll
        for (int ni = 0; ni < 4; ++ni) {
            int col = blockN + wn * 64 + ni * 16 + l16;
            #pragma unroll
            for (int j = 0; j < 8; ++j) {
                int row = blockM + wm * 32 + mi * 16 + half * 8 + j;
                H[(size_t)row * NCOL + col] = acc[mi][ni][j];
            }
        }
    }
}

// ---------------------------------------------------------------- attention scalars
__global__ void alpha_kernel(const float* __restrict__ h,
                             const float* __restrict__ a_src,
                             const float* __restrict__ a_dst,
                             float* __restrict__ asrc, float* __restrict__ adst,
                             int nRows) {
    int idx = blockIdx.x * blockDim.x + threadIdx.x;
    if (idx >= nRows * HEADS) return;
    int n = idx >> 3, hd = idx & 7;
    const float* hp = h + (size_t)n * NCOL + hd * HID;
    const float* as = a_src + hd * HID;
    const float* ad = a_dst + hd * HID;
    float s0 = 0.f, s1 = 0.f;
    #pragma unroll
    for (int c = 0; c < HID; ++c) { s0 += hp[c] * as[c]; s1 += hp[c] * ad[c]; }
    asrc[idx] = s0;
    adst[idx] = s1;
}

__global__ void fill_f32(float* __restrict__ p, float v, long long n) {
    long long i = (long long)blockIdx.x * blockDim.x + threadIdx.x;
    long long stride = (long long)gridDim.x * blockDim.x;
    for (; i < n; i += stride) p[i] = v;
}

// ---------------------------------------------------------------- edge passes
__global__ void edge_max_kernel(const int* __restrict__ src, const int* __restrict__ dst,
                                int E, int nRows,
                                const float* __restrict__ asrc, const float* __restrict__ adst,
                                float* __restrict__ m) {
    int idx = blockIdx.x * blockDim.x + threadIdx.x;
    int total = (E + nRows) * HEADS;
    if (idx >= total) return;
    int e = idx >> 3, hd = idx & 7;
    int s, d;
    if (e < E) { s = src[e]; d = dst[e]; } else { s = d = e - E; }
    float ev = asrc[s * HEADS + hd] + adst[d * HEADS + hd];
    ev = ev > 0.f ? ev : 0.2f * ev;                     // leaky_relu
    atomicMaxF(&m[d * HEADS + hd], ev);
}

__global__ void edge_acc_kernel(const int* __restrict__ src, const int* __restrict__ dst,
                                int E, int nRows,
                                const float* __restrict__ asrc, const float* __restrict__ adst,
                                const float* __restrict__ m,
                                const float* __restrict__ h,
                                float* __restrict__ denom, float* __restrict__ accum) {
    int idx = blockIdx.x * blockDim.x + threadIdx.x;
    int total = (E + nRows) * HEADS;
    if (idx >= total) return;
    int e = idx >> 3, hd = idx & 7;
    int s, d;
    if (e < E) { s = src[e]; d = dst[e]; } else { s = d = e - E; }
    float ev = asrc[s * HEADS + hd] + adst[d * HEADS + hd];
    ev = ev > 0.f ? ev : 0.2f * ev;
    float ex = __expf(ev - m[d * HEADS + hd]);
    atomicAdd(&denom[d * HEADS + hd], ex);
    const float4* hp = (const float4*)(h + (size_t)s * NCOL + hd * HID);
    float* op = accum + (size_t)d * NCOL + hd * HID;
    #pragma unroll
    for (int i = 0; i < HID / 4; ++i) {
        float4 v = hp[i];
        atomicAdd(op + i * 4 + 0, ex * v.x);
        atomicAdd(op + i * 4 + 1, ex * v.y);
        atomicAdd(op + i * 4 + 2, ex * v.z);
        atomicAdd(op + i * 4 + 3, ex * v.w);
    }
}

// normalize softmax, add bias, ELU; write fp32 in place + bf16 copy (next GEMM input)
__global__ void node_fin_kernel(float* __restrict__ accum,
                                const float* __restrict__ denom,
                                const float* __restrict__ bias,
                                unsigned short* __restrict__ outbf,
                                int nRows, long long total) {
    long long idx = (long long)blockIdx.x * blockDim.x + threadIdx.x;
    if (idx >= total) return;
    int n = (int)(idx >> 8), c = (int)(idx & (NCOL - 1));
    if (n < nRows) {
        float v = accum[idx] / (denom[n * HEADS + (c >> 5)] + 1e-16f) + bias[c];
        v = v > 0.f ? v : (__expf(v) - 1.0f);           // ELU(alpha=1)
        accum[idx] = v;
        outbf[idx] = f2bf(v);
    } else {
        outbf[idx] = 0;                                  // zero pad rows
    }
}

// ---------------------------------------------------------------- pooling + head
__global__ void pool_kernel(const float* __restrict__ hfin, const int* __restrict__ batch,
                            float* __restrict__ pooled, float* __restrict__ cnt, int nRows) {
    long long idx = (long long)blockIdx.x * blockDim.x + threadIdx.x;
    if (idx >= (long long)nRows * NCOL) return;
    int n = (int)(idx >> 8), c = (int)(idx & (NCOL - 1));
    int b = batch[n];
    atomicAdd(&pooled[(size_t)b * NCOL + c], hfin[idx]);
    if (c == 0) atomicAdd(&cnt[b], 1.0f);
}

__global__ void final_kernel(const float* __restrict__ pooled, const float* __restrict__ cnt,
                             const float* __restrict__ lin_w, const float* __restrict__ lin_b,
                             float* __restrict__ out, int G) {
    int g = blockIdx.x * blockDim.x + threadIdx.x;
    if (g >= G) return;
    float cg = fmaxf(cnt[g], 1.0f);
    const float* pp = pooled + (size_t)g * NCOL;
    float s = 0.f;
    #pragma unroll 4
    for (int c = 0; c < NCOL; ++c) s += pp[c] * lin_w[c];
    out[g] = s / cg + lin_b[0];
}

// ---------------------------------------------------------------- launch
extern "C" void kernel_launch(void* const* d_in, const int* in_sizes, int n_in,
                              void* d_out, int out_size, void* d_ws, size_t ws_size,
                              hipStream_t stream) {
    const float* x     = (const float*)d_in[0];
    const int*   eidx  = (const int*)d_in[1];
    const int*   batch = (const int*)d_in[2];
    const float* Wl[3]    = {(const float*)d_in[3],  (const float*)d_in[7],  (const float*)d_in[11]};
    const float* aSrcl[3] = {(const float*)d_in[4],  (const float*)d_in[8],  (const float*)d_in[12]};
    const float* aDstl[3] = {(const float*)d_in[5],  (const float*)d_in[9],  (const float*)d_in[13]};
    const float* biasl[3] = {(const float*)d_in[6],  (const float*)d_in[10], (const float*)d_in[14]};
    const float* lin_w = (const float*)d_in[15];
    const float* lin_b = (const float*)d_in[16];
    float*       out   = (float*)d_out;

    const int Nn   = in_sizes[0] / 128;         // 50000
    const int E    = in_sizes[1] / 2;           // 800000
    const int G    = out_size;                  // 512
    const int Ep   = E + Nn;                    // with self loops
    const int Mpad = ((Nn + 127) / 128) * 128;  // 50048

    // workspace layout
    float* ws     = (float*)d_ws;
    float* hbuf   = ws;                               // Mpad*256  (GEMM out, fp32)
    float* accum  = hbuf  + (size_t)Mpad * NCOL;      // Mpad*256  (msg accum / next fp32 input)
    float* asrc   = accum + (size_t)Mpad * NCOL;      // N*8
    float* adst   = asrc  + (size_t)Nn * HEADS;       // N*8
    float* mmax   = adst  + (size_t)Nn * HEADS;       // N*8
    float* denom  = mmax  + (size_t)Nn * HEADS;       // N*8
    float* pooled = denom + (size_t)Nn * HEADS;       // G*256
    float* cnt    = pooled + (size_t)G * NCOL;        // G
    unsigned short* inbf = (unsigned short*)(cnt + G);          // Mpad*256 bf16 (GEMM A)
    unsigned short* wbfT = inbf + (size_t)Mpad * NCOL;          // 256*256 bf16 (GEMM B, [n][k])

    const int* srcP = eidx;
    const int* dstP = eidx + E;

    const dim3 blk(256);
    const int nhBlocks   = (Nn * HEADS + 255) / 256;
    const int edgeBlocks = (Ep * HEADS + 255) / 256;
    const int ncBlocks   = (int)(((long long)Nn * NCOL + 255) / 256);
    const int npBlocks   = (int)(((long long)Mpad * NCOL + 255) / 256);

    for (int l = 0; l < 3; ++l) {
        const int K      = (l == 0) ? 128 : NCOL;
        const int kshift = (l == 0) ? 7 : 8;

        if (l == 0) {
            long long tot = (long long)Mpad * 128;
            convert_in_kernel<<<(int)((tot + 255) / 256), blk, 0, stream>>>(x, inbf, Nn, 7, tot);
        }
        convert_wT_kernel<<<(NCOL * K + 255) / 256, blk, 0, stream>>>(Wl[l], wbfT, kshift, NCOL * K);

        gat_gemm_wmma<<<dim3(Mpad / 128, NCOL / 128), blk, 0, stream>>>(inbf, wbfT, hbuf, K);

        alpha_kernel<<<nhBlocks, blk, 0, stream>>>(hbuf, aSrcl[l], aDstl[l], asrc, adst, Nn);
        fill_f32<<<1024, blk, 0, stream>>>(mmax, -INFINITY, (long long)Nn * HEADS);
        fill_f32<<<1024, blk, 0, stream>>>(denom, 0.0f, (long long)Nn * HEADS);
        fill_f32<<<4096, blk, 0, stream>>>(accum, 0.0f, (long long)Nn * NCOL);
        edge_max_kernel<<<edgeBlocks, blk, 0, stream>>>(srcP, dstP, E, Nn, asrc, adst, mmax);
        edge_acc_kernel<<<edgeBlocks, blk, 0, stream>>>(srcP, dstP, E, Nn, asrc, adst,
                                                        mmax, hbuf, denom, accum);
        node_fin_kernel<<<npBlocks, blk, 0, stream>>>(accum, denom, biasl[l], inbf, Nn,
                                                      (long long)Mpad * NCOL);
    }

    fill_f32<<<256, blk, 0, stream>>>(pooled, 0.0f, (long long)G * NCOL);
    fill_f32<<<8, blk, 0, stream>>>(cnt, 0.0f, (long long)G);
    pool_kernel<<<ncBlocks, blk, 0, stream>>>(accum, batch, pooled, cnt, Nn);
    final_kernel<<<(G + 255) / 256, blk, 0, stream>>>(pooled, cnt, lin_w, lin_b, out, G);
}
